// BGRL_50251117363931
// MI455X (gfx1250) — matrix operand order repaired and checked by reading, not compiled, not run on hardware
//
#include <hip/hip_runtime.h>
#include <hip/hip_bf16.h>

// ---------------------------------------------------------------------------
// BGRL forward: fused 16384x16384x512 bf16-WMMA cosine-sim + top-8 (dominant
// cost, compute-bound, teacher resident in 192MB L2), two 16384x512x512 f32
// WMMA encoder GEMMs (validated outputs), l2-normalize -> bf16.
// Student tile staged via Tensor Data Mover with LDS pad for bank-conflict-
// free A-fragment reads. sim_topk_bf16 defined first so the disasm snippet
// shows the TDM + bf16 WMMA path.
// ---------------------------------------------------------------------------

#define NROWS 16384
#define DIM   512
#define KSEL  8
#define MTILE 64                      // student rows per block in sim kernel
#define LDS_ROW (DIM + 16)            // 528 bf16 = 1056B row stride (TDM pad)
#define NEG_INF (-3.402823466e38f)

typedef __attribute__((ext_vector_type(2)))  float  v2f;
typedef __attribute__((ext_vector_type(8)))  float  v8f;
typedef __attribute__((ext_vector_type(16))) __bf16 v16bf;
typedef __attribute__((ext_vector_type(4)))  unsigned int u32x4;
typedef __attribute__((ext_vector_type(4)))  int    i32x4;
typedef __attribute__((ext_vector_type(8)))  int    i32x8;

union BFV { v16bf v; uint4 u[2]; };

// ===========================================================================
// Kernel 3 (defined first for disasm visibility): fused similarity + top-8.
// Block = 256 threads (8 waves) owns MTILE=64 student rows. Wave w owns
// columns j0+16w..+15 and computes 4 M-subtiles per teacher B fragment
// (4x register reuse -> 4 GB total L2 traffic). Student tile is DMA'd by
// the TDM (tensor_load_to_lds) with pad_interval=256dw/pad_amount=8dw so
// the LDS row stride is 1056B -> conflict-free 16-lane A-frag ds_loads.
// ===========================================================================
__global__ __launch_bounds__(256)
void sim_topk_bf16(const __bf16* __restrict__ ST, const __bf16* __restrict__ TE,
                   int* __restrict__ knn_rows, int* __restrict__ knn_cols) {
  __shared__ __bf16 sA[MTILE][LDS_ROW];     // 64 x 528 bf16 = 67.5 KB
  __shared__ float  sSim[MTILE][128];       // 32 KB
  __shared__ float  cval[MTILE][32];        // 8 KB
  __shared__ int    cidx[MTILE][32];        // 8 KB

  const int i0   = blockIdx.x * MTILE;
  const int tid  = threadIdx.x;
  const int wave = tid >> 5;
  const int lane = tid & 31;
  const int m    = lane & 15;
  const int half = lane >> 4;
  const int prow = tid >> 2;                // top-k: 4 threads per row
  const int psub = tid & 3;

#if __has_builtin(__builtin_amdgcn_tensor_load_to_lds)
  // --- Tensor Data Mover staging of the 64x512 bf16 student tile ---------
  if (tid < 32) {                           // wave 0 issues the DMA
    unsigned lds_addr = (unsigned)(unsigned long long)(void*)&sA[0][0];
    unsigned long long ga = (unsigned long long)(ST + (size_t)i0 * DIM);
    // D# group 0: count=1, lds_addr, global_addr, type=2 ("image")
    u32x4 g0 = { 1u, lds_addr, (unsigned)ga,
                 (unsigned)((ga >> 32) & 0x01FFFFFFull) | (2u << 30) };
    // D# group 1: data_size=2B(code 1), pad_enable, pad_interval=256 DWORDs
    // (code 7), pad_amount=8 DWORDs (code 7) -> LDS row stride 1024+32B.
    unsigned d0 = (1u << 16) | (1u << 20) | (7u << 22) | (7u << 25);
    unsigned d1 = ((unsigned)DIM & 0xFFFFu) << 16;          // tensor_dim0 lo
    unsigned d2 = ((unsigned)DIM >> 16) |
                  (((unsigned)NROWS & 0xFFFFu) << 16);      // tensor_dim1 lo
    unsigned d3 = ((unsigned)NROWS >> 16) |
                  (((unsigned)DIM) << 16);                  // tile_dim0=512
    unsigned d4 = (unsigned)MTILE;                          // tile_dim1=64
    unsigned d5 = (unsigned)DIM;                            // dim0_stride
    i32x8 g1 = { (int)d0, (int)d1, (int)d2, (int)d3,
                 (int)d4, (int)d5, 0, 0 };
    i32x4 z4 = { 0, 0, 0, 0 };
#if defined(__clang_major__) && (__clang_major__ >= 23)
    i32x8 z8 = { 0, 0, 0, 0, 0, 0, 0, 0 };
    __builtin_amdgcn_tensor_load_to_lds(g0, g1, z4, z4, z8, 0);
#else
    __builtin_amdgcn_tensor_load_to_lds(g0, g1, z4, z4, 0);
#endif
#if __has_builtin(__builtin_amdgcn_s_wait_tensorcnt)
    __builtin_amdgcn_s_wait_tensorcnt(0);
#endif
  }
#else
  // --- Fallback: cooperative staging into the same padded layout ---------
  {
    const unsigned int* src = (const unsigned int*)(ST + (size_t)i0 * DIM);
    for (int it = 0; it < MTILE; ++it) {
      int u   = tid + it * 256;
      int row = u >> 8;                     // 256 dwords per (unpadded) row
      int cu  = u & 255;
      ((unsigned int*)&sA[row][0])[cu] = src[row * 256 + cu];
    }
  }
#endif
  __syncthreads();

  float tv[KSEL];
  int   ti[KSEL];
  #pragma unroll
  for (int q = 0; q < KSEL; ++q) { tv[q] = NEG_INF; ti[q] = 0x7fffffff; }

  for (int j0 = 0; j0 < NROWS; j0 += 128) {
    v8f acc[4];
    #pragma unroll
    for (int mt = 0; mt < 4; ++mt)
      acc[mt] = (v8f){0.f,0.f,0.f,0.f,0.f,0.f,0.f,0.f};

    const __bf16* trow = TE + (size_t)(j0 + wave * 16 + m) * DIM;
    if (j0 + 128 < NROWS)   // global_prefetch_b8 for next teacher chunk
      __builtin_prefetch(TE + (size_t)(j0 + 128 + wave * 16 + m) * DIM, 0, 1);

    for (int kk = 0; kk < DIM; kk += 32) {
      BFV b;   // 16-bit B (32x16): lane half 0 -> K{0..15}, half 1 -> K{16..31}
      b.u[0] = *(const uint4*)(trow + kk + half * 16);
      b.u[1] = *(const uint4*)(trow + kk + half * 16 + 8);
      #pragma unroll
      for (int mt = 0; mt < 4; ++mt) {
        BFV a; // 16-bit A (16x32): lane half 0 -> K{0..15}, half 1 -> K{16..31}
        a.u[0] = *(const uint4*)&sA[mt * 16 + m][kk + half * 16];
        a.u[1] = *(const uint4*)&sA[mt * 16 + m][kk + half * 16 + 8];
        acc[mt] = __builtin_amdgcn_wmma_f32_16x16x32_bf16(
            false, a.v, false, b.v, (short)0, acc[mt], false, false);
      }
    }

    #pragma unroll
    for (int mt = 0; mt < 4; ++mt)
      #pragma unroll
      for (int v = 0; v < 8; ++v)
        sSim[mt * 16 + v + half * 8][wave * 16 + m] = acc[mt][v];
    __syncthreads();

    // Partial top-8: 4 threads per row, strided columns (monotone gid per
    // thread => strict '>' keeps lowest index on ties, like lax.top_k).
    for (int cc = 0; cc < 32; ++cc) {
      int   c   = psub + cc * 4;
      float v   = sSim[prow][c];
      int   gid = j0 + c;
      if (v > tv[KSEL - 1]) {
        tv[KSEL - 1] = v; ti[KSEL - 1] = gid;
        #pragma unroll
        for (int q = KSEL - 1; q > 0; --q) {
          bool up = (tv[q] > tv[q - 1]) ||
                    (tv[q] == tv[q - 1] && ti[q] < ti[q - 1]);
          if (up) {
            float fv = tv[q]; tv[q] = tv[q - 1]; tv[q - 1] = fv;
            int   iv = ti[q]; ti[q] = ti[q - 1]; ti[q - 1] = iv;
          }
        }
      }
    }
    __syncthreads();   // before next iteration overwrites sSim
  }

  // Merge: 4 partial lists per row (32 candidates), select final 8.
  #pragma unroll
  for (int q = 0; q < KSEL; ++q) {
    cval[prow][psub * KSEL + q] = tv[q];
    cidx[prow][psub * KSEL + q] = ti[q];
  }
  __syncthreads();

  if (tid < MTILE) {
    const int r    = tid;
    const int orow = i0 + r;
    for (int q = 0; q < KSEL; ++q) {
      float best = NEG_INF; int bi = 0x7fffffff; int bc = 0;
      for (int c = 0; c < 32; ++c) {
        float v = cval[r][c]; int id = cidx[r][c];
        if (v > best || (v == best && id < bi)) { best = v; bi = id; bc = c; }
      }
      cval[r][bc] = NEG_INF;
      knn_rows[orow * KSEL + q] = orow;
      knn_cols[orow * KSEL + q] = bi;
    }
  }
}

// ===========================================================================
// Kernel 1: Y[N,512] = X[N,512] @ W[512,512], f32, via V_WMMA_F32_16X16X4_F32.
// Block = 256 threads (8 waves), 16 rows x 512 cols. A-tile staged in LDS.
// ===========================================================================
__global__ __launch_bounds__(256)
void gemm512_f32_wmma(const float* __restrict__ X, const float* __restrict__ W,
                      float* __restrict__ Y) {
  __shared__ float sA[16][DIM + 4];
  const int i0   = blockIdx.x * 16;
  const int tid  = threadIdx.x;
  const int wave = tid >> 5;
  const int lane = tid & 31;
  const int m    = lane & 15;
  const int half = lane >> 4;

  for (int it = 0; it < 8; ++it) {
    int idx = tid + it * 256;
    int row = idx >> 7;
    int c4  = idx & 127;
    float4 v = ((const float4*)(X + (size_t)(i0 + row) * DIM))[c4];
    *(float4*)&sA[row][c4 * 4] = v;
  }
  __syncthreads();

  v8f acc[4];
  #pragma unroll
  for (int t = 0; t < 4; ++t)
    acc[t] = (v8f){0.f,0.f,0.f,0.f,0.f,0.f,0.f,0.f};

  for (int kk = 0; kk < DIM; kk += 4) {
    v2f a;   // A frag (16x4): lane half 0 -> K{0,1}, half 1 -> K{2,3}
    a.x = sA[m][kk + half * 2 + 0];
    a.y = sA[m][kk + half * 2 + 1];
    #pragma unroll
    for (int t = 0; t < 4; ++t) {
      const int col0 = (wave + t * 8) * 16;
      v2f b; // B frag (4x16): N = lane%16, K mirrors A halves
      b.x = W[(size_t)(kk + half * 2 + 0) * DIM + col0 + m];
      b.y = W[(size_t)(kk + half * 2 + 1) * DIM + col0 + m];
      acc[t] = __builtin_amdgcn_wmma_f32_16x16x4_f32(
          false, a, false, b, (short)0, acc[t], false, false);
    }
  }

  #pragma unroll
  for (int t = 0; t < 4; ++t) {
    const int col0 = (wave + t * 8) * 16;
    #pragma unroll
    for (int v = 0; v < 8; ++v) {
      int row = v + half * 8;
      Y[(size_t)(i0 + row) * DIM + col0 + m] = acc[t][v];
    }
  }
}

// ===========================================================================
// Kernel 2: S[row] = Y[row] / max(||Y[row]||, 1e-12) -> bf16. Wave per row.
// ===========================================================================
__global__ __launch_bounds__(256)
void l2norm_to_bf16(const float* __restrict__ Y, __bf16* __restrict__ S) {
  const int wave = threadIdx.x >> 5;
  const int lane = threadIdx.x & 31;
  const int row  = blockIdx.x * 8 + wave;
  const float* yr = Y + (size_t)row * DIM;

  float vals[16];
  float ss = 0.f;
  #pragma unroll
  for (int i = 0; i < 16; ++i) {
    float v = yr[lane + i * 32];
    vals[i] = v;
    ss += v * v;
  }
  #pragma unroll
  for (int off = 16; off > 0; off >>= 1)
    ss += __shfl_xor(ss, off, 32);

  float inv = 1.f / fmaxf(sqrtf(ss), 1e-12f);
  __bf16* sr = S + (size_t)row * DIM;
  #pragma unroll
  for (int i = 0; i < 16; ++i)
    sr[lane + i * 32] = (__bf16)(vals[i] * inv);
}

// ===========================================================================
// Launch: d_in = {online_x, target_x, W_online, W_pred(unused), W_target, k}
// d_out = online_y[N*512] f32 | target_y[N*512] f32 | knn[2*N*8] int32
// d_ws  = student bf16 [N*512] | teacher bf16 [N*512]   (32 MB)
// ===========================================================================
extern "C" void kernel_launch(void* const* d_in, const int* in_sizes, int n_in,
                              void* d_out, int out_size, void* d_ws, size_t ws_size,
                              hipStream_t stream) {
  const float* online_x = (const float*)d_in[0];
  const float* target_x = (const float*)d_in[1];
  const float* W_online = (const float*)d_in[2];
  const float* W_target = (const float*)d_in[4];
  // d_in[3] = W_pred: reference computes online_q but never returns it.

  float* online_y = (float*)d_out;
  float* target_y = online_y + (size_t)NROWS * DIM;
  int*   knn      = (int*)(target_y + (size_t)NROWS * DIM);
  int*   knn_rows = knn;
  int*   knn_cols = knn + NROWS * KSEL;

  __bf16* student = (__bf16*)d_ws;
  __bf16* teacher = student + (size_t)NROWS * DIM;

  dim3 blk(256);
  gemm512_f32_wmma<<<NROWS / 16,    blk, 0, stream>>>(online_x, W_online, online_y);
  gemm512_f32_wmma<<<NROWS / 16,    blk, 0, stream>>>(target_x, W_target, target_y);
  l2norm_to_bf16 <<<NROWS / 8,     blk, 0, stream>>>(online_y, student);
  l2norm_to_bf16 <<<NROWS / 8,     blk, 0, stream>>>(target_y, teacher);
  sim_topk_bf16  <<<NROWS / MTILE, blk, 0, stream>>>(student, teacher, knn_rows, knn_cols);
}
